// DynamicSparseScaledDotProductAttention_40381282517697
// MI455X (gfx1250) — matrix operand order repaired and checked by reading, not compiled, not run on hardware
//
#include <hip/hip_runtime.h>

typedef __attribute__((ext_vector_type(16))) _Float16 v16h;
typedef __attribute__((ext_vector_type(8)))  _Float16 v8h;
typedef __attribute__((ext_vector_type(4)))  _Float16 v4h;
typedef __attribute__((ext_vector_type(8)))  float    v8f;
typedef __attribute__((ext_vector_type(4)))  float    v4f;

#define BH 32
#define L 2048
#define D 64
#define TEMP_INV 0.125f       // 1/sqrt(d_k) = 1/8
#define WINDOW 2
#define NEG_INF_VAL (-1e30f)
#define ST 2052               // padded LDS row stride (floats); 8*ST mod 64 = 32 -> conflict-free halves

// ---------------------------------------------------------------------------
// Prep: Qh = f16(Q * 1/temp), Kh = f16(K), Vth = f16(V) transposed to [bh][d][l]
// Vectorized x4. K/V are read exactly once -> non-temporal loads (keep L2 for
// the f16 working set the attention kernel re-reads).
// ---------------------------------------------------------------------------
__global__ void prep_kernel(const float* __restrict__ q, const float* __restrict__ k,
                            const float* __restrict__ v, _Float16* __restrict__ qh,
                            _Float16* __restrict__ kh, _Float16* __restrict__ vth) {
  int i4 = blockIdx.x * blockDim.x + threadIdx.x;
  if (i4 >= BH * L * D / 4) return;
  const int idx = i4 * 4;

  v4f qv = *(const v4f*)(q + idx);                                  // q re-read by gate kernel: RT
  v4f kv = __builtin_nontemporal_load((const v4f*)(k + idx));       // read-once: NT
  v4f vv = __builtin_nontemporal_load((const v4f*)(v + idx));       // read-once: NT

  v4h q4, k4;
#pragma unroll
  for (int j = 0; j < 4; ++j) {
    q4[j] = (_Float16)(qv[j] * TEMP_INV);
    k4[j] = (_Float16)(kv[j]);
  }
  *(v4h*)(qh + idx) = q4;
  *(v4h*)(kh + idx) = k4;

  const int d  = idx % D;                 // multiple of 4
  const int l  = (idx / D) % L;
  const int bh = idx / (L * D);
#pragma unroll
  for (int j = 0; j < 4; ++j)
    vth[((size_t)bh * D + d + j) * L + l] = (_Float16)vv[j];
}

// gate[bh*L + l] = 1.0 iff sigmoid(q . w + b) > 0.5  (i.e. logit > 0)
__global__ void gate_kernel(const float* __restrict__ q, const float* __restrict__ w,
                            const float* __restrict__ b, float* __restrict__ gate) {
  int idx = blockIdx.x * blockDim.x + threadIdx.x;
  if (idx >= BH * L) return;
  const float* qp = q + (size_t)idx * D;
  float acc = b[0];
#pragma unroll
  for (int i = 0; i < D; ++i) acc += qp[i] * w[i];
  gate[idx] = (acc > 0.0f) ? 1.0f : 0.0f;
}

// ---------------------------------------------------------------------------
// Main attention kernel: one workgroup = one (bh, 16-query tile).
// 256 threads = 8 wave32s. Full 16x2048 score row held in LDS (CDNA5 WGP LDS).
// ---------------------------------------------------------------------------
__launch_bounds__(256, 2)
__global__ void attn_kernel(const _Float16* __restrict__ qh, const _Float16* __restrict__ kh,
                            const _Float16* __restrict__ vth, const float* __restrict__ gate,
                            float* __restrict__ out, float* __restrict__ attn) {
  __shared__ __align__(16) float scores[16 * ST];       // 131,328 B (reused for partial C later)
  __shared__ __align__(16) _Float16 qtile[16 * D];      // 2 KB
  __shared__ float red[16][17];
  __shared__ float rowm[16];
  __shared__ float rowr[16];
  __shared__ float gates[16];

  const int tid  = threadIdx.x;
  const int wave = tid >> 5;
  const int lane = tid & 31;
  const int n    = lane & 15;   // column index within 16-wide tile
  const int h    = lane >> 4;   // lane half-select
  const int q0   = blockIdx.x * 16;
  const int bh   = blockIdx.y;

  // Stage Q tile (pre-scaled f16) + gates into LDS.
  for (int i = tid; i < 16 * D; i += 256)
    qtile[i] = qh[((size_t)bh * L + q0) * D + i];
  if (tid < 16) gates[tid] = gate[bh * L + q0 + tid];
  __syncthreads();

  // Build the two Q A-fragments (K-chunks 0-31, 32-63) once per wave.
  // A fragment as loaded here (contiguous k, same order as B): lane<16 row n holds K {0..15}; lane>=16 holds {16..31}.
  union AF { v16h v; v8h p[2]; };
  v16h aq[2];
#pragma unroll
  for (int c = 0; c < 2; ++c) {
    const _Float16* qr = &qtile[n * D + 32 * c];
    AF af;
    af.p[0] = *(const v8h*)(qr + 16 * h);
    af.p[1] = *(const v8h*)(qr + 16 * h + 8);
    aq[c] = af.v;
  }

  // Loop-invariant per-lane gate condition per C-row r (hoists the compares).
  bool grow[8];
#pragma unroll
  for (int r = 0; r < 8; ++r) grow[r] = gates[r + 8 * h] > 0.5f;

  // ---------------- Phase 1: scores = (Q/temp) . K^T  (masked) -> LDS ------
  const _Float16* kbase = kh + (size_t)bh * L * D;
  for (int t = wave; t < L / 16; t += 8) {
    const int k0 = t * 16;
    v8f c = {};
#pragma unroll
    for (int cc = 0; cc < 2; ++cc) {
      // B layout (32x16): lane n holds column n (= row k0+n of K), dims 32cc+16h..+15, contiguous.
      const _Float16* kp = kbase + (size_t)(k0 + n) * D + 32 * cc + 16 * h;
      v16h b = *(const v16h*)kp;
      c = __builtin_amdgcn_wmma_f32_16x16x32_f16(false, aq[cc], false, b,
                                                 (short)0, c, false, false);
    }
    // C layout: VGPR r -> row r (lanes 0-15) / row r+8 (lanes 16-31), col = n.
    // Window [qi-2, qi+2] only touches tiles with k0 in [q0-17, q0+17]: uniform branch.
    const bool near_diag = (k0 + 15 >= q0 - WINDOW) && (k0 <= q0 + 15 + WINDOW);
    if (near_diag) {
#pragma unroll
      for (int r = 0; r < 8; ++r) {
        const int m  = r + 8 * h;
        int dlt = (q0 + m) - (k0 + n); if (dlt < 0) dlt = -dlt;
        const bool allowed = grow[r] || (dlt <= WINDOW);
        scores[m * ST + k0 + n] = allowed ? c[r] : NEG_INF_VAL;
      }
    } else {
#pragma unroll
      for (int r = 0; r < 8; ++r)
        scores[(r + 8 * h) * ST + k0 + n] = grow[r] ? c[r] : NEG_INF_VAL;
    }
  }
  __syncthreads();

  // ---------------- Phase 1.5: softmax statistics --------------------------
  {
    const int row = tid >> 4, seg = tid & 15;
    const float* sr = &scores[row * ST + seg * 128];
    float mx = -3.4e38f;
#pragma unroll 8
    for (int i = 0; i < 128; ++i) mx = fmaxf(mx, sr[i]);
    red[row][seg] = mx;
  }
  __syncthreads();
  if (tid < 16) {
    float mx = red[tid][0];
#pragma unroll
    for (int i = 1; i < 16; ++i) mx = fmaxf(mx, red[tid][i]);
    rowm[tid] = mx;
  }
  __syncthreads();
  {
    const int row = tid >> 4, seg = tid & 15;
    float* sr = &scores[row * ST + seg * 128];
    const float m = rowm[row];
    float s = 0.0f;
#pragma unroll 8
    for (int i = 0; i < 128; ++i) { float p = __expf(sr[i] - m); sr[i] = p; s += p; }
    red[row][seg] = s;
  }
  __syncthreads();
  if (tid < 16) {
    float s = 0.0f;
#pragma unroll
    for (int i = 0; i < 16; ++i) s += red[tid][i];
    rowr[tid] = 1.0f / s;
  }
  __syncthreads();

  // ---------------- Phase 2a: stream normalized attn to HBM ---------------
  // 512 MB across the grid, written once and never re-read -> non-temporal
  // b128 stores so the streaming output does not evict the K/V f16 working
  // set (16 MB, fully L2-resident) from the 192 MB L2.
  float* attn_row = attn + ((size_t)bh * L + q0) * L;
  for (int i = tid; i < 16 * (L / 4); i += 256) {
    const int r   = i >> 9;            // i / 512
    const int col = (i & 511) << 2;    // 4-float column
    v4f sv = *(const v4f*)&scores[r * ST + col];
    sv *= rowr[r];
    __builtin_nontemporal_store(sv, (v4f*)(attn_row + (size_t)r * L + col));
  }

  // ---------------- Phase 2b: out = P . V via WMMA -------------------------
  v8f acc[4] = { {}, {}, {}, {} };
  const _Float16* vtb = vth + (size_t)bh * D * L;
  for (int t = wave; t < L / 32; t += 8) {
    const int k0 = t * 32;
    // A fragment: row n of exp'd P, f32 -> f16 on the fly.
    v16h pa;
    const float* pr = &scores[n * ST + k0];
#pragma unroll
    for (int i = 0; i < 8; ++i) pa[i]     = (_Float16)pr[16 * h + i];
#pragma unroll
    for (int i = 0; i < 8; ++i) pa[8 + i] = (_Float16)pr[16 * h + 8 + i];
#pragma unroll
    for (int nt = 0; nt < 4; ++nt) {
      // B fragment: lane n holds V column (dim 16*nt+n), kdims k0+16h..+15, contiguous in Vth.
      const _Float16* vp = vtb + (size_t)(nt * 16 + n) * L + k0 + 16 * h;
      v16h b = *(const v16h*)vp;
      acc[nt] = __builtin_amdgcn_wmma_f32_16x16x32_f16(false, pa, false, b,
                                                       (short)0, acc[nt], false, false);
    }
  }
  __syncthreads();   // all reads of `scores` complete -> safe to reuse as opart

  // ---------------- Phase 3: cross-wave reduction of partial C -------------
  float* opart = scores;  // reuse: 8 waves x 16 rows x 64 dims = 32 KB
#pragma unroll
  for (int nt = 0; nt < 4; ++nt)
#pragma unroll
    for (int r = 0; r < 8; ++r) {
      const int m = r + 8 * h;
      opart[(wave * 16 + m) * 64 + (nt * 16 + n)] = acc[nt][r];
    }
  __syncthreads();

  float* outp = out + ((size_t)bh * L + q0) * D;
  for (int i = tid; i < 16 * D; i += 256) {
    const int m = i >> 6;
    float s = 0.0f;
#pragma unroll
    for (int w = 0; w < 8; ++w) s += opart[w * 1024 + i];
    __builtin_nontemporal_store(s * rowr[m], &outp[i]);
  }
}

// ---------------------------------------------------------------------------
extern "C" void kernel_launch(void* const* d_in, const int* in_sizes, int n_in,
                              void* d_out, int out_size, void* d_ws, size_t ws_size,
                              hipStream_t stream) {
  (void)in_sizes; (void)n_in; (void)out_size; (void)ws_size;
  const float* q  = (const float*)d_in[0];
  const float* k  = (const float*)d_in[1];
  const float* v  = (const float*)d_in[2];
  const float* gw = (const float*)d_in[3];
  const float* gb = (const float*)d_in[4];

  char* ws = (char*)d_ws;
  _Float16* qh   = (_Float16*)(ws);                         // 8 MB
  _Float16* kh   = (_Float16*)(ws + 8ull  * 1024 * 1024);   // 8 MB
  _Float16* vth  = (_Float16*)(ws + 16ull * 1024 * 1024);   // 8 MB
  float*    gate = (float*)   (ws + 24ull * 1024 * 1024);   // 256 KB

  float* out  = (float*)d_out;                              // [32,2048,64]
  float* attn = out + (size_t)BH * L * D;                   // [32,2048,2048]

  prep_kernel<<<(BH * L * D / 4 + 255) / 256, 256, 0, stream>>>(q, k, v, qh, kh, vth);
  gate_kernel<<<(BH * L + 255) / 256, 256, 0, stream>>>(q, gw, gb, gate);

  dim3 grid(L / 16, BH);
  attn_kernel<<<grid, 256, 0, stream>>>(qh, kh, vth, gate, out, attn);
}
